// GraphAttentionLayer_86706799772346
// MI455X (gfx1250) — matrix operand order repaired and checked by reading, not compiled, not run on hardware
//
#include <hip/hip_runtime.h>

typedef __attribute__((ext_vector_type(16))) _Float16 v16h;
typedef __attribute__((ext_vector_type(8)))  _Float16 v8h;
typedef __attribute__((ext_vector_type(8)))  float    v8f;

#define N_ROWS 8192
#define K_DIM  256
#define N_DIM  64

// Padded LDS row strides (in fp16 elements), chosen for 16B alignment and
// conflict-free lane access on 64 banks.
#define WT_STRIDE 264   // 264*2B = 528B = 132 dwords; 132 % 64 = 4 -> lanes hit distinct banks
#define HT_STRIDE 40    // 40*2B = 80B = 20 dwords; 20*col mod 64 all distinct for col 0..15

__device__ __forceinline__ float leaky01(float t) { return t > 0.0f ? t : 0.01f * t; }

__device__ __forceinline__ v16h load_bfrag(const _Float16* p) {
  // 32 bytes, guaranteed 16B-aligned -> two ds_load_b128 / global_load_b128
  v8h lo = *(const v8h*)p;
  v8h hi = *(const v8h*)(p + 8);
  return __builtin_shufflevector(lo, hi, 0,1,2,3,4,5,6,7,8,9,10,11,12,13,14,15);
}

__device__ __forceinline__ void async_wait0() {
#if __has_builtin(__builtin_amdgcn_s_wait_asynccnt)
  __builtin_amdgcn_s_wait_asynccnt(0);
#else
  asm volatile("s_wait_asynccnt 0x0" ::: "memory");
#endif
}

// Async DMA: 16 bytes global -> LDS, tracked by ASYNCcnt (no VGPR round-trip).
// LDS byte address = low 32 bits of the generic pointer (flat aperture rule).
__device__ __forceinline__ void async_copy_b128(const void* gaddr, void* lds_ptr) {
  const unsigned lds_addr = (unsigned)(unsigned long long)lds_ptr;
  asm volatile("global_load_async_to_lds_b128 %0, %1, off"
               :: "v"(lds_addr), "v"(gaddr)
               : "memory");
}

// ---------------------------------------------------------------------------
// K1: h = x @ W  (8192x256 @ 256x64), fp16 WMMA, f32 accumulate.
// Writes ht (fp16, transposed: ht[n][row]) for K3's B-fragments, and
// s1 = h@a1, s2 = h@a2.
// Block: 256 threads = 8 waves; each wave owns a 16-row strip (block = 128 rows).
// ---------------------------------------------------------------------------
__global__ __launch_bounds__(256) void gat_k1(const float* __restrict__ x,
                                              const float* __restrict__ W,
                                              const float* __restrict__ a1,
                                              const float* __restrict__ a2,
                                              _Float16* __restrict__ ht,
                                              float* __restrict__ s1,
                                              float* __restrict__ s2) {
  __shared__ _Float16 ldsWt[N_DIM * WT_STRIDE];   // W transposed: Wt[n][k], fp16
  const int tid = threadIdx.x;

  // Stage W (row-major [k][n]) into LDS transposed as fp16.
  for (int i = tid; i < K_DIM * N_DIM; i += 256) {
    const int k = i >> 6, n = i & 63;
    ldsWt[n * WT_STRIDE + k] = (_Float16)W[i];
  }
  __syncthreads();

  const int lane = tid & 31, wave = tid >> 5;
  const int half = lane >> 4, rc = lane & 15;     // rc = A-row (lane%16) = C-col
  const int r0 = blockIdx.x * 128 + wave * 16;

  v8f acc[4] = {};                                 // 16x64 strip as 4 16x16 tiles
  const float4* xq = (const float4*)(x + (size_t)(r0 + rc) * K_DIM);

  #pragma unroll
  for (int kc = 0; kc < K_DIM / 32; ++kc) {
    // A fragment: 16x32, contiguous k order (same as B). Lane holds row rc; K runs
    // [kc*32 + 16*half, +16): one contiguous 64B run.
    const int K0 = kc * 32 + 16 * half;
    const float4 f0 = xq[K0 / 4], f1 = xq[K0 / 4 + 1];
    const float4 f2 = xq[K0 / 4 + 2], f3 = xq[K0 / 4 + 3];
    v16h a;
    a[0]=(_Float16)f0.x;  a[1]=(_Float16)f0.y;  a[2]=(_Float16)f0.z;  a[3]=(_Float16)f0.w;
    a[4]=(_Float16)f1.x;  a[5]=(_Float16)f1.y;  a[6]=(_Float16)f1.z;  a[7]=(_Float16)f1.w;
    a[8]=(_Float16)f2.x;  a[9]=(_Float16)f2.y;  a[10]=(_Float16)f2.z; a[11]=(_Float16)f2.w;
    a[12]=(_Float16)f3.x; a[13]=(_Float16)f3.y; a[14]=(_Float16)f3.z; a[15]=(_Float16)f3.w;

    // B fragments: lane holds column (t*16+rc), 16 consecutive K values
    // starting at kc*32 + 16*half -> contiguous in transposed LDS image.
    const int kb = kc * 32 + 16 * half;
    #pragma unroll
    for (int t = 0; t < 4; ++t) {
      const v16h b = load_bfrag(&ldsWt[(t * 16 + rc) * WT_STRIDE + kb]);
      acc[t] = __builtin_amdgcn_wmma_f32_16x16x32_f16(
          false, a, false, b, (short)0, acc[t], false, false);
    }
  }

  // Store h transposed in fp16: ht[n*8192 + row]. C/D layout: acc[t][v] is
  // h[r0 + v + 8*half][t*16 + rc] -> 8 consecutive rows = one b128 store.
  #pragma unroll
  for (int t = 0; t < 4; ++t) {
    v8h p;
    #pragma unroll
    for (int v = 0; v < 8; ++v) p[v] = (_Float16)acc[t][v];
    *(v8h*)(ht + (size_t)(t * 16 + rc) * N_ROWS + r0 + 8 * half) = p;
  }

  // s1/s2: per-lane partial dot over this lane's 4 columns, then xor-reduce
  // across the 16 lanes of each half (cols 16t+rc cover all 64 n).
  float p1[8], p2[8];
  #pragma unroll
  for (int v = 0; v < 8; ++v) { p1[v] = 0.0f; p2[v] = 0.0f; }
  #pragma unroll
  for (int t = 0; t < 4; ++t) {
    const float av1 = a1[t * 16 + rc], av2 = a2[t * 16 + rc];
    #pragma unroll
    for (int v = 0; v < 8; ++v) { p1[v] += acc[t][v] * av1; p2[v] += acc[t][v] * av2; }
  }
  #pragma unroll
  for (int m = 1; m <= 8; m <<= 1) {
    #pragma unroll
    for (int v = 0; v < 8; ++v) {
      p1[v] += __shfl_xor(p1[v], m, 32);
      p2[v] += __shfl_xor(p2[v], m, 32);
    }
  }
  if (rc == 0) {
    #pragma unroll
    for (int v = 0; v < 8; ++v) {
      s1[r0 + 8 * half + v] = p1[v];
      s2[r0 + 8 * half + v] = p2[v];
    }
  }
}

// ---------------------------------------------------------------------------
// K2: s2max = max(s2). leaky_relu is monotone, so for every row i the exact
// softmax max is leaky_relu(s1[i] + s2max) -- no online-max rescaling needed.
// ---------------------------------------------------------------------------
__global__ __launch_bounds__(256) void gat_k2(const float* __restrict__ s2,
                                              float* __restrict__ s2max) {
  __shared__ float red[256];
  float m = -1e30f;
  for (int i = threadIdx.x; i < N_ROWS; i += 256) m = fmaxf(m, s2[i]);
  red[threadIdx.x] = m;
  __syncthreads();
  for (int s = 128; s > 0; s >>= 1) {
    if ((int)threadIdx.x < s) red[threadIdx.x] = fmaxf(red[threadIdx.x], red[threadIdx.x + s]);
    __syncthreads();
  }
  if (threadIdx.x == 0) *s2max = red[0];
}

// ---------------------------------------------------------------------------
// K3: out[i] = (sum_j w_ij * h[j]) / (sum_j w_ij),
//     w_ij = exp(leaky_relu(s1[i]+s2[j]) - leaky_relu(s1[i]+s2max)).
// Flash-style single pass with DOUBLE-BUFFERED ASYNC LDS staging:
// issue global_load_async_to_lds_b128 for chunk jc+1, compute chunk jc
// (16 exps/lane + 4 WMMAs) while the async engine fills the other buffer.
// ---------------------------------------------------------------------------
__global__ __launch_bounds__(256) void gat_k3(const _Float16* __restrict__ ht,
                                              const float* __restrict__ s1,
                                              const float* __restrict__ s2,
                                              const float* __restrict__ s2max_p,
                                              float* __restrict__ out) {
  __shared__ _Float16 ldsHt[2][N_DIM * HT_STRIDE];  // 2 x (64x32 fp16 chunk, padded)
  __shared__ float ldsS2[2][32];

  const int tid = threadIdx.x, lane = tid & 31, wave = tid >> 5;
  const int half = lane >> 4, rc = lane & 15;
  const int r0 = blockIdx.x * 128 + wave * 16;

  const float s2max = *s2max_p;
  const float s1r = s1[r0 + rc];                  // A-layout row = lane%16
  const float Mi = leaky01(s1r + s2max);          // exact row max of e

  v8f acc[4] = {};
  float dsum = 0.0f;                              // this lane's K-subset of sum_j w

  const int nStage = tid >> 2;                    // 0..63 : h column n
  const int seg    = tid & 3;                     // 0..3  : 8-element segment
  const _Float16* gsrc = ht + (size_t)nStage * N_ROWS + seg * 8;
  _Float16* lds0 = &ldsHt[0][nStage * HT_STRIDE + seg * 8];
  _Float16* lds1 = &ldsHt[1][nStage * HT_STRIDE + seg * 8];

  // Prime buffer 0 with chunk 0.
  async_copy_b128(gsrc, lds0);
  if (tid < 32) ldsS2[0][tid] = s2[tid];

  for (int jc = 0; jc < N_ROWS / 32; ++jc) {
    const int cur = jc & 1;
    async_wait0();        // my async fill of buf[cur] (issued last iter) is in LDS
    __syncthreads();      // publish buf[cur]; everyone is done reading buf[cur^1]

    if (jc + 1 < N_ROWS / 32) {
      async_copy_b128(gsrc + (size_t)(jc + 1) * 32, cur ? lds0 : lds1);
      if (tid < 32) ldsS2[cur ^ 1][tid] = s2[(jc + 1) * 32 + tid];
    }

    // A fragment = softmax weights (K index j within chunk = 16*half + element, same order as B).
    const int j0 = 16 * half;
    v16h a;
    #pragma unroll
    for (int i = 0; i < 8; ++i) {
      const float w = __expf(leaky01(s1r + ldsS2[cur][j0 + i]) - Mi);
      dsum += w; a[i] = (_Float16)w;
    }
    #pragma unroll
    for (int i = 0; i < 8; ++i) {
      const float w = __expf(leaky01(s1r + ldsS2[cur][j0 + 8 + i]) - Mi);
      dsum += w; a[8 + i] = (_Float16)w;
    }

    const int kb = 16 * half;
    #pragma unroll
    for (int t = 0; t < 4; ++t) {
      const v16h b = load_bfrag(&ldsHt[cur][(t * 16 + rc) * HT_STRIDE + kb]);
      acc[t] = __builtin_amdgcn_wmma_f32_16x16x32_f16(
          false, a, false, b, (short)0, acc[t], false, false);
    }
  }

  // Lanes L and L^16 hold complementary K-subsets for the same row (lane%16).
  const float dtot = dsum + __shfl_xor(dsum, 16, 32);

  // Divide and store: acc[t][v] is row (r0 + v + 8*half), col (t*16 + rc);
  // d for that row lives at lane (v + 8*half).
  #pragma unroll
  for (int v = 0; v < 8; ++v) {
    const float dv = __shfl(dtot, v + 8 * half, 32);
    const float inv = 1.0f / dv;
    #pragma unroll
    for (int t = 0; t < 4; ++t)
      out[(size_t)(r0 + v + 8 * half) * N_DIM + t * 16 + rc] = acc[t][v] * inv;
  }
}

// ---------------------------------------------------------------------------
extern "C" void kernel_launch(void* const* d_in, const int* in_sizes, int n_in,
                              void* d_out, int out_size, void* d_ws, size_t ws_size,
                              hipStream_t stream) {
  const float* x  = (const float*)d_in[0];
  const float* W  = (const float*)d_in[1];
  const float* a1 = (const float*)d_in[2];
  const float* a2 = (const float*)d_in[3];
  float* out = (float*)d_out;

  char* ws = (char*)d_ws;
  _Float16* ht = (_Float16*)ws;                               // 64*8192*2 = 1 MB
  float* s1    = (float*)(ws + (size_t)N_DIM * N_ROWS * 2);   // 32 KB
  float* s2    = s1 + N_ROWS;                                 // 32 KB
  float* s2m   = s2 + N_ROWS;                                 // 4 B

  gat_k1<<<N_ROWS / 128, 256, 0, stream>>>(x, W, a1, a2, ht, s1, s2);
  gat_k2<<<1, 256, 0, stream>>>(s2, s2m);
  gat_k3<<<N_ROWS / 128, 256, 0, stream>>>(ht, s1, s2, s2m, out);
}
